// GemmRSTritonNVLink_86646670229698
// MI455X (gfx1250) — matrix operand order repaired and checked by reading, not compiled, not run on hardware
//
#include <hip/hip_runtime.h>

// C[4096,4096] = sum_r x[r] @ w[r]  (reduce-scatter reshape is a layout no-op)
// Two-phase: (1) fp32 -> bf16 hi/lo split (+ B transpose) into workspace,
// (2) WMMA GEMM with bf16x3 accumulation, async global->LDS staging.

namespace {

constexpr int MDIM   = 4096;
constexpr int NDIM   = 4096;
constexpr int KP     = 1024;  // K per rank
constexpr int WRANKS = 8;
constexpr int BM = 128, BN = 128, BK = 32;
constexpr int LDA    = BK + 8;                 // 40 bf16 = 80 B padded LDS stride
constexpr int KSTEPS = WRANKS * (KP / BK);     // 256

constexpr size_t SHARD_ELEMS = (size_t)WRANKS * MDIM * KP;   // 33,554,432
constexpr size_t SHARD_BYTES = SHARD_ELEMS * 2;              // 64 MiB (bf16)
constexpr size_t WS_NEEDED   = 4 * SHARD_BYTES;              // 256 MiB

typedef __attribute__((ext_vector_type(16))) __bf16 bf16x16;
typedef __attribute__((ext_vector_type(8)))  float  floatx8;

struct alignas(16) U4 { unsigned x, y, z, w; };
struct alignas(16) F4 { float x, y, z, w; };
union FragAB { bf16x16 v; U4 q[2]; };
union Pack4  { unsigned short h[4]; unsigned long long u; };

__device__ __forceinline__ unsigned short f32_to_bf16_rne(float f) {
  unsigned u = __float_as_uint(f);
  u += 0x7FFFu + ((u >> 16) & 1u);   // RNE; inputs finite
  return (unsigned short)(u >> 16);
}
__device__ __forceinline__ float bf16_to_f32(unsigned short h) {
  return __uint_as_float((unsigned)h << 16);
}

// CDNA5 async copy: 16 B per lane, global -> LDS, tracked by ASYNCcnt.
__device__ __forceinline__ void async_copy_b128(unsigned ldsAddr, const void* gaddr) {
  asm volatile("global_load_async_to_lds_b128 %0, %1, off"
               :: "v"(ldsAddr), "v"(gaddr) : "memory");
}
__device__ __forceinline__ void wait_async() {
  asm volatile("s_wait_asynccnt 0" ::: "memory");
}
__device__ __forceinline__ unsigned lds_addr(const void* p) {
  return (unsigned)(size_t)p;   // low 32 bits of generic shared ptr = LDS offset
}

} // namespace

// ---------------------------------------------------------------------------
// Pass 1a: elementwise split of x: f32 [r][m][k] -> bf16 hi/lo, same layout.
// ---------------------------------------------------------------------------
__global__ __launch_bounds__(256)
void split_x_kernel(const float* __restrict__ in,
                    unsigned short* __restrict__ hi,
                    unsigned short* __restrict__ lo) {
  const size_t i = ((size_t)blockIdx.x * 256 + threadIdx.x) * 4;
  const F4 v = *(const F4*)(in + i);
  const float vv[4] = {v.x, v.y, v.z, v.w};
  Pack4 ph, pl;
#pragma unroll
  for (int j = 0; j < 4; ++j) {
    const unsigned short hb = f32_to_bf16_rne(vv[j]);
    ph.h[j] = hb;
    pl.h[j] = f32_to_bf16_rne(vv[j] - bf16_to_f32(hb));
  }
  *(unsigned long long*)(hi + i) = ph.u;
  *(unsigned long long*)(lo + i) = pl.u;
}

// ---------------------------------------------------------------------------
// Pass 1b: split + transpose of w: f32 [r][k][n] -> bf16 hi/lo [r][n][k].
// 64x64 tiles through LDS so both global sides stay coalesced.
// ---------------------------------------------------------------------------
__global__ __launch_bounds__(256)
void split_w_t_kernel(const float* __restrict__ in,
                      unsigned short* __restrict__ hiT,
                      unsigned short* __restrict__ loT) {
  __shared__ unsigned short sh[64][72];
  __shared__ unsigned short sl[64][72];
  const int r   = blockIdx.z;
  const int bk  = blockIdx.y * 64;
  const int bn_ = blockIdx.x * 64;
  const int t   = threadIdx.x;

  const int col4 = (t & 15) * 4;   // n within tile
  const int rowk = t >> 4;         // 0..15
#pragma unroll
  for (int p = 0; p < 4; ++p) {
    const int k = rowk + 16 * p;
    const F4 v = *(const F4*)(in + ((size_t)r * KP + (bk + k)) * NDIM + bn_ + col4);
    const float vv[4] = {v.x, v.y, v.z, v.w};
#pragma unroll
    for (int j = 0; j < 4; ++j) {
      const unsigned short hb = f32_to_bf16_rne(vv[j]);
      sh[col4 + j][k] = hb;
      sl[col4 + j][k] = f32_to_bf16_rne(vv[j] - bf16_to_f32(hb));
    }
  }
  __syncthreads();

  const int k4   = (t & 15) * 4;
  const int rown = t >> 4;
#pragma unroll
  for (int p = 0; p < 4; ++p) {
    const int n = rown + 16 * p;
    Pack4 ph, pl;
#pragma unroll
    for (int j = 0; j < 4; ++j) { ph.h[j] = sh[n][k4 + j]; pl.h[j] = sl[n][k4 + j]; }
    const size_t o = ((size_t)r * NDIM + (bn_ + n)) * KP + bk + k4;
    *(unsigned long long*)(hiT + o) = ph.u;
    *(unsigned long long*)(loT + o) = pl.u;
  }
}

// ---------------------------------------------------------------------------
// Pass 2: GEMM, bf16x3, async global->LDS staging (no VALU in the data path).
// ---------------------------------------------------------------------------
__global__ __launch_bounds__(256)
void gemm_wmma_pre(const unsigned short* __restrict__ xh,
                   const unsigned short* __restrict__ xl,
                   const unsigned short* __restrict__ whT,  // [r][n][k]
                   const unsigned short* __restrict__ wlT,
                   float* __restrict__ out) {
  __shared__ unsigned short sAh[2][BM * LDA];
  __shared__ unsigned short sAl[2][BM * LDA];
  __shared__ unsigned short sBh[2][BN * LDA];
  __shared__ unsigned short sBl[2][BN * LDA];

  const int tid   = threadIdx.x;
  const int lane  = tid & 31;
  const int wave  = tid >> 5;
  const int waveM = (wave & 3) * 32;
  const int waveN = (wave >> 2) * 64;
  const int bm = blockIdx.y * BM;
  const int bn = blockIdx.x * BN;

  const int fRow  = lane & 15;
  const int aKsel = (lane >> 4) << 3;   // A frag: K {0..7,16..23} / {8..15,24..31}
  const int bKsel = (lane >> 4) << 3;   // B frag: K {0..7,16..23} / {8..15,24..31}

  floatx8 acc[2][4];
#pragma unroll
  for (int tm = 0; tm < 2; ++tm)
#pragma unroll
    for (int tn = 0; tn < 4; ++tn)
#pragma unroll
      for (int j = 0; j < 8; ++j) acc[tm][tn][j] = 0.0f;

  // Stage both 128x32 bf16 panels (hi+lo) for step kt into LDS buffer `buf`.
  // 512 16-B chunks per array; 256 threads -> 2 chunks each -> 8 asyncs/thread.
  auto asyncStage = [&](int buf, int kt) {
    const int r  = kt >> 5;
    const int kb = (kt & 31) * BK;
#pragma unroll
    for (int p = 0; p < 2; ++p) {
      const int c   = tid + 256 * p;
      const int row = c >> 2;         // 0..127
      const int kof = (c & 3) * 8;    // elem offset of 16-B chunk
      const int l   = row * LDA + kof;
      const size_t aIdx = ((size_t)r * MDIM + (bm + row)) * KP + kb + kof;
      async_copy_b128(lds_addr(&sAh[buf][l]), xh + aIdx);
      async_copy_b128(lds_addr(&sAl[buf][l]), xl + aIdx);
      const size_t bIdx = ((size_t)r * NDIM + (bn + row)) * KP + kb + kof;
      async_copy_b128(lds_addr(&sBh[buf][l]), whT + bIdx);
      async_copy_b128(lds_addr(&sBl[buf][l]), wlT + bIdx);
    }
  };

  auto computeTile = [&](int buf) {
    FragAB ah[2], al[2], bh[4], bl[4];
#pragma unroll
    for (int tm = 0; tm < 2; ++tm) {
      const int off = (waveM + tm * 16 + fRow) * LDA + aKsel;
      ah[tm].q[0] = *(const U4*)&sAh[buf][off];
      ah[tm].q[1] = *(const U4*)&sAh[buf][off + 16];
      al[tm].q[0] = *(const U4*)&sAl[buf][off];
      al[tm].q[1] = *(const U4*)&sAl[buf][off + 16];
    }
#pragma unroll
    for (int tn = 0; tn < 4; ++tn) {
      const int off = (waveN + tn * 16 + fRow) * LDA + bKsel;
      bh[tn].q[0] = *(const U4*)&sBh[buf][off];
      bh[tn].q[1] = *(const U4*)&sBh[buf][off + 16];
      bl[tn].q[0] = *(const U4*)&sBl[buf][off];
      bl[tn].q[1] = *(const U4*)&sBl[buf][off + 16];
    }
#pragma unroll
    for (int tm = 0; tm < 2; ++tm)
#pragma unroll
      for (int tn = 0; tn < 4; ++tn) {
        acc[tm][tn] = __builtin_amdgcn_wmma_f32_16x16x32_bf16(
            false, ah[tm].v, false, bh[tn].v, (short)0, acc[tm][tn], false, false);
        acc[tm][tn] = __builtin_amdgcn_wmma_f32_16x16x32_bf16(
            false, ah[tm].v, false, bl[tn].v, (short)0, acc[tm][tn], false, false);
        acc[tm][tn] = __builtin_amdgcn_wmma_f32_16x16x32_bf16(
            false, al[tm].v, false, bh[tn].v, (short)0, acc[tm][tn], false, false);
      }
  };

  asyncStage(0, 0);
  wait_async();
  __syncthreads();

  for (int kt = 0; kt < KSTEPS; ++kt) {
    const int  cur  = kt & 1;
    const bool more = (kt + 1) < KSTEPS;
    if (more) asyncStage(cur ^ 1, kt + 1);  // fire-and-forget into other buffer
    computeTile(cur);                        // ds_load_b128 + 24x v_wmma
    if (more) wait_async();                  // drain ASYNCcnt for next buffer
    __syncthreads();
  }

  const int rowSel = (lane >> 4) << 3;
#pragma unroll
  for (int tm = 0; tm < 2; ++tm)
#pragma unroll
    for (int tn = 0; tn < 4; ++tn) {
      const int col  = bn + waveN + tn * 16 + fRow;
      const int row0 = bm + waveM + tm * 16 + rowSel;
#pragma unroll
      for (int j = 0; j < 8; ++j)
        out[(size_t)(row0 + j) * NDIM + col] = acc[tm][tn][j];
    }
}

// ---------------------------------------------------------------------------
// Fallback: fully fused kernel (used when ws_size < 256 MiB).
// ---------------------------------------------------------------------------
__global__ __launch_bounds__(256)
void gemm_rs_wmma_fused(const float* __restrict__ x,
                        const float* __restrict__ w,
                        float* __restrict__ out) {
  __shared__ unsigned short sAh[2][BM * LDA];
  __shared__ unsigned short sAl[2][BM * LDA];
  __shared__ unsigned short sBh[2][BN * LDA];
  __shared__ unsigned short sBl[2][BN * LDA];

  const int tid   = threadIdx.x;
  const int lane  = tid & 31;
  const int wave  = tid >> 5;
  const int waveM = (wave & 3) * 32;
  const int waveN = (wave >> 2) * 64;
  const int bm = blockIdx.y * BM;
  const int bn = blockIdx.x * BN;

  const int aRow = tid >> 3;
  const int aCol = (tid & 7) << 2;
  const int bCol = tid & 127;
  const int bK4  = (tid >> 7) << 2;

  const int fRow  = lane & 15;
  const int aKsel = (lane >> 4) << 3;
  const int bKsel = (lane >> 4) << 3;

  F4    aReg[4];
  float bReg[4][4];

  floatx8 acc[2][4];
#pragma unroll
  for (int tm = 0; tm < 2; ++tm)
#pragma unroll
    for (int tn = 0; tn < 4; ++tn)
#pragma unroll
      for (int j = 0; j < 8; ++j) acc[tm][tn][j] = 0.0f;

  auto loadGlobal = [&](int kt) {
    const int r  = kt >> 5;
    const int kb = (kt & 31) * BK;
    const float* pa = x + ((size_t)r * MDIM + (size_t)(bm + aRow)) * KP + (kb + aCol);
#pragma unroll
    for (int p = 0; p < 4; ++p)
      aReg[p] = *(const F4*)(pa + (size_t)p * 32 * KP);
    const float* pb = w + ((size_t)r * KP + (size_t)(kb + bK4)) * NDIM + (bn + bCol);
#pragma unroll
    for (int p = 0; p < 4; ++p)
#pragma unroll
      for (int j = 0; j < 4; ++j)
        bReg[p][j] = pb[((size_t)p * 8 + j) * NDIM];
  };

  auto stageLds = [&](int buf) {
#pragma unroll
    for (int p = 0; p < 4; ++p) {
      const float v[4] = {aReg[p].x, aReg[p].y, aReg[p].z, aReg[p].w};
      Pack4 ph, pl;
#pragma unroll
      for (int j = 0; j < 4; ++j) {
        const unsigned short hb = f32_to_bf16_rne(v[j]);
        ph.h[j] = hb;
        pl.h[j] = f32_to_bf16_rne(v[j] - bf16_to_f32(hb));
      }
      const int off = (aRow + 32 * p) * LDA + aCol;
      *(unsigned long long*)&sAh[buf][off] = ph.u;
      *(unsigned long long*)&sAl[buf][off] = pl.u;
    }
#pragma unroll
    for (int p = 0; p < 4; ++p) {
      Pack4 ph, pl;
#pragma unroll
      for (int j = 0; j < 4; ++j) {
        const unsigned short hb = f32_to_bf16_rne(bReg[p][j]);
        ph.h[j] = hb;
        pl.h[j] = f32_to_bf16_rne(bReg[p][j] - bf16_to_f32(hb));
      }
      const int off = bCol * LDA + (p * 8 + bK4);
      *(unsigned long long*)&sBh[buf][off] = ph.u;
      *(unsigned long long*)&sBl[buf][off] = pl.u;
    }
  };

  auto computeTile = [&](int buf) {
    FragAB ah[2], al[2], bh[4], bl[4];
#pragma unroll
    for (int tm = 0; tm < 2; ++tm) {
      const int off = (waveM + tm * 16 + fRow) * LDA + aKsel;
      ah[tm].q[0] = *(const U4*)&sAh[buf][off];
      ah[tm].q[1] = *(const U4*)&sAh[buf][off + 16];
      al[tm].q[0] = *(const U4*)&sAl[buf][off];
      al[tm].q[1] = *(const U4*)&sAl[buf][off + 16];
    }
#pragma unroll
    for (int tn = 0; tn < 4; ++tn) {
      const int off = (waveN + tn * 16 + fRow) * LDA + bKsel;
      bh[tn].q[0] = *(const U4*)&sBh[buf][off];
      bh[tn].q[1] = *(const U4*)&sBh[buf][off + 16];
      bl[tn].q[0] = *(const U4*)&sBl[buf][off];
      bl[tn].q[1] = *(const U4*)&sBl[buf][off + 16];
    }
#pragma unroll
    for (int tm = 0; tm < 2; ++tm)
#pragma unroll
      for (int tn = 0; tn < 4; ++tn) {
        acc[tm][tn] = __builtin_amdgcn_wmma_f32_16x16x32_bf16(
            false, ah[tm].v, false, bh[tn].v, (short)0, acc[tm][tn], false, false);
        acc[tm][tn] = __builtin_amdgcn_wmma_f32_16x16x32_bf16(
            false, ah[tm].v, false, bl[tn].v, (short)0, acc[tm][tn], false, false);
        acc[tm][tn] = __builtin_amdgcn_wmma_f32_16x16x32_bf16(
            false, al[tm].v, false, bh[tn].v, (short)0, acc[tm][tn], false, false);
      }
  };

  loadGlobal(0);
  stageLds(0);
  __syncthreads();

  for (int kt = 0; kt < KSTEPS; ++kt) {
    const int  cur  = kt & 1;
    const bool more = (kt + 1) < KSTEPS;
    if (more) loadGlobal(kt + 1);
    computeTile(cur);
    if (more) stageLds(cur ^ 1);
    __syncthreads();
  }

  const int rowSel = (lane >> 4) << 3;
#pragma unroll
  for (int tm = 0; tm < 2; ++tm)
#pragma unroll
    for (int tn = 0; tn < 4; ++tn) {
      const int col  = bn + waveN + tn * 16 + fRow;
      const int row0 = bm + waveM + tm * 16 + rowSel;
#pragma unroll
      for (int j = 0; j < 8; ++j)
        out[(size_t)(row0 + j) * NDIM + col] = acc[tm][tn][j];
    }
}

extern "C" void kernel_launch(void* const* d_in, const int* in_sizes, int n_in,
                              void* d_out, int out_size, void* d_ws, size_t ws_size,
                              hipStream_t stream) {
  (void)in_sizes; (void)n_in; (void)out_size;
  const float* x = (const float*)d_in[0];   // [W, M, KP] f32
  const float* w = (const float*)d_in[1];   // [W, KP, N] f32
  float* out = (float*)d_out;               // [M, N] f32
  const dim3 grid(NDIM / BN, MDIM / BM, 1);

  if (ws_size >= WS_NEEDED && d_ws != nullptr) {
    unsigned short* xh  = (unsigned short*)d_ws;
    unsigned short* xl  = xh + SHARD_ELEMS;
    unsigned short* whT = xl + SHARD_ELEMS;
    unsigned short* wlT = whT + SHARD_ELEMS;

    split_x_kernel<<<dim3((unsigned)(SHARD_ELEMS / (256 * 4))), dim3(256), 0, stream>>>(
        x, xh, xl);
    split_w_t_kernel<<<dim3(NDIM / 64, KP / 64, WRANKS), dim3(256), 0, stream>>>(
        w, whT, wlT);
    gemm_wmma_pre<<<grid, dim3(256), 0, stream>>>(xh, xl, whT, wlT, out);
  } else {
    gemm_rs_wmma_fused<<<grid, dim3(256), 0, stream>>>(x, w, out);
  }
}
